// preGATConv_4784593568247
// MI455X (gfx1250) — matrix operand order repaired and checked by reading, not compiled, not run on hardware
//
#include <hip/hip_runtime.h>
#include <limits.h>

#define N_NODES 50000
#define E_EDGES 800000
#define IN_F    256
#define OUT_F   128
#define NEG_SLOPE 0.2f
#define ALPHA     0.5f

typedef __attribute__((ext_vector_type(16))) __bf16 v16bf;
typedef __attribute__((ext_vector_type(8)))  __bf16 v8bf;
typedef __attribute__((ext_vector_type(8)))  float  v8f;

union BF16x16 { v16bf v; v8bf h[2]; };

// fp32 -> bf16 round-to-nearest-even
static __device__ __forceinline__ unsigned short f2bf(float f) {
    unsigned int u = __float_as_uint(f);
    unsigned int r = u + 0x7FFFu + ((u >> 16) & 1u);
    return (unsigned short)(r >> 16);
}

// two fp32 -> packed bf16x2 in one dword
static __device__ __forceinline__ unsigned int pk2bf(float a, float b) {
    return (unsigned int)f2bf(a) | ((unsigned int)f2bf(b) << 16);
}

// order-preserving float <-> int mapping for integer atomicMax
static __device__ __forceinline__ int fenc(float f) {
    int i = __float_as_int(f);
    return i >= 0 ? i : (i ^ 0x7FFFFFFF);
}
static __device__ __forceinline__ float fdec(int k) {
    return __int_as_float(k >= 0 ? k : (k ^ 0x7FFFFFFF));
}

static __device__ __forceinline__ float lrelu(float e) {
    return e > 0.0f ? e : NEG_SLOPE * e;
}

// ---------------------------------------------------------------------------
// Kernel 0: init. out = broadcast(bias); per-node max/sum accumulators reset.
// ---------------------------------------------------------------------------
__global__ __launch_bounds__(256) void gat_init(const float* __restrict__ bias,
                                                float* __restrict__ out,
                                                int* __restrict__ maxE,
                                                int* __restrict__ maxW,
                                                float* __restrict__ sumE,
                                                float* __restrict__ sumW) {
    size_t i = (size_t)blockIdx.x * 256 + threadIdx.x;
    if (i < (size_t)N_NODES * OUT_F) out[i] = bias[i & (OUT_F - 1)];
    if (i < N_NODES) {
        maxE[i] = INT_MIN;
        maxW[i] = INT_MIN;
        sumE[i] = 0.0f;
        sumW[i] = 0.0f;
    }
}

// ---------------------------------------------------------------------------
// Kernel 1: ft = feat @ fc_w via v_wmma_f32_16x16x32_bf16 (f32 accumulate),
// fused epilogue computes el = ft.attn_l, er = ft.attn_r per row.
// One 256-thread block (8 waves) -> 128x128 output tile; K-loop of 8 x 32.
// A tile staged as packed bf16x2 dwords (ds_store_b64 per float4 chunk);
// next K-slice of feat is global_prefetch'ed to overlap HBM latency.
// ---------------------------------------------------------------------------
__global__ __launch_bounds__(256) void gat_gemm(const float* __restrict__ feat,
                                                const float* __restrict__ fcw,
                                                const float* __restrict__ attn_l,
                                                const float* __restrict__ attn_r,
                                                float* __restrict__ ft,
                                                float* __restrict__ el,
                                                float* __restrict__ er) {
    __shared__ unsigned short As[128][32];  // [row][k]   bf16, 8 KB
    __shared__ unsigned short Bt[128][32];  // [col][k]   bf16, 8 KB (B transposed)
    __shared__ float sAL[OUT_F];
    __shared__ float sAR[OUT_F];

    const int t = threadIdx.x;
    if (t < OUT_F) sAL[t] = attn_l[t];
    else           sAR[t - OUT_F] = attn_r[t - OUT_F];

    const int rb   = blockIdx.x * 128;
    const int wv   = t >> 5;
    const int lane = t & 31;
    const int half = lane >> 4;
    const int lrow = lane & 15;

    v8f acc[8] = {};  // 8 N-subtiles of 16 cols each

    for (int k0 = 0; k0 < IN_F; k0 += 32) {
        // ---- stage A tile: feat[rb..rb+127][k0..k0+31] -> packed bf16 LDS ----
        #pragma unroll
        for (int j = 0; j < 4; ++j) {
            int u   = t + 256 * j;        // float4 unit, 1024 total
            int row = u >> 3;
            int c4  = u & 7;
            const bool inb = (rb + row) < N_NODES;
            const float* p = feat + (size_t)(rb + row) * IN_F + k0 + c4 * 4;
            float4 v = make_float4(0.f, 0.f, 0.f, 0.f);
            if (inb) {
                v = *reinterpret_cast<const float4*>(p);
                if (k0 + 32 < IN_F) __builtin_prefetch(p + 32, 0, 1);  // next K slice
            }
            uint2 pk;
            pk.x = pk2bf(v.x, v.y);
            pk.y = pk2bf(v.z, v.w);
            *reinterpret_cast<uint2*>(&As[row][c4 * 4]) = pk;  // one ds_store_b64
        }
        // ---- stage B tile (transposed): fc_w[k0..k0+31][0..127] ----
        #pragma unroll
        for (int j = 0; j < 4; ++j) {
            int u  = t + 256 * j;
            int kk = u >> 5;              // 0..31
            int c4 = u & 31;              // 0..31 (float4 units across 128 cols)
            float4 v = *reinterpret_cast<const float4*>(fcw + (size_t)(k0 + kk) * OUT_F + c4 * 4);
            Bt[c4 * 4 + 0][kk] = f2bf(v.x);
            Bt[c4 * 4 + 1][kk] = f2bf(v.y);
            Bt[c4 * 4 + 2][kk] = f2bf(v.z);
            Bt[c4 * 4 + 3][kk] = f2bf(v.w);
        }
        __syncthreads();

        // A fragment (ISA 16-bit 16x32 layout): lanes 0-15 rows M=lrow with
        // K {0..7,16..23}; lanes 16-31 same rows with K {8..15,24..31}.
        const int arow = wv * 16 + lrow;
        BF16x16 afrag;
        afrag.h[0] = *reinterpret_cast<const v8bf*>(&As[arow][half ? 8 : 0]);
        afrag.h[1] = *reinterpret_cast<const v8bf*>(&As[arow][half ? 24 : 16]);

        #pragma unroll
        for (int n = 0; n < 8; ++n) {
            // B fragment: lane holds column n*16+lrow, K {0..7,16..23} (half 0) / {8..15,24..31} (half 1)
            const int col = n * 16 + lrow;
            BF16x16 bfrag;
            bfrag.h[0] = *reinterpret_cast<const v8bf*>(&Bt[col][half * 8]);
            bfrag.h[1] = *reinterpret_cast<const v8bf*>(&Bt[col][16 + half * 8]);
            acc[n] = __builtin_amdgcn_wmma_f32_16x16x32_bf16(
                false, afrag.v, false, bfrag.v, (short)0, acc[n], false, false);
        }
        __syncthreads();
    }

    // ---- epilogue: store ft tile, fused el/er dot-products ----
    float pl[8] = {0}, pr[8] = {0};
    #pragma unroll
    for (int n = 0; n < 8; ++n) {
        const int col = n * 16 + lrow;
        const float al = sAL[col], ar = sAR[col];
        #pragma unroll
        for (int r = 0; r < 8; ++r) {
            float v = acc[n][r];
            int row = rb + wv * 16 + half * 8 + r;   // C/D layout: lane<16 -> M=r, else M=r+8
            if (row < N_NODES) ft[(size_t)row * OUT_F + col] = v;
            pl[r] += v * al;
            pr[r] += v * ar;
        }
    }
    // reduce across the 16 lanes of each half-wave (masks stay within halves)
    #pragma unroll
    for (int r = 0; r < 8; ++r) {
        float a = pl[r], b = pr[r];
        #pragma unroll
        for (int m = 1; m < 16; m <<= 1) {
            a += __shfl_xor(a, m, 32);
            b += __shfl_xor(b, m, 32);
        }
        int row = rb + wv * 16 + half * 8 + r;
        if (lrow == 0 && row < N_NODES) { el[row] = a; er[row] = b; }
    }
}

// ---------------------------------------------------------------------------
// Kernel 2: per-edge leaky-relu score; segment max over dst for both e and w
// via order-preserving integer atomicMax.
// ---------------------------------------------------------------------------
__global__ __launch_bounds__(256) void gat_edge_max(const float* __restrict__ el,
                                                    const float* __restrict__ er,
                                                    const float* __restrict__ w,
                                                    const int* __restrict__ src,
                                                    const int* __restrict__ dst,
                                                    int* __restrict__ maxE,
                                                    int* __restrict__ maxW) {
    int i = blockIdx.x * 256 + threadIdx.x;
    if (i >= E_EDGES) return;
    int s = src[i], d = dst[i];
    float e = lrelu(el[s] + er[d]);
    atomicMax(&maxE[d], fenc(e));
    atomicMax(&maxW[d], fenc(w[i]));
}

// ---------------------------------------------------------------------------
// Kernel 3: exp(x - max) partial sums per destination (both softmaxes).
// e is recomputed from el/er (cheaper than an E-sized spill buffer).
// ---------------------------------------------------------------------------
__global__ __launch_bounds__(256) void gat_edge_sum(const float* __restrict__ el,
                                                    const float* __restrict__ er,
                                                    const float* __restrict__ w,
                                                    const int* __restrict__ src,
                                                    const int* __restrict__ dst,
                                                    const int* __restrict__ maxE,
                                                    const int* __restrict__ maxW,
                                                    float* __restrict__ sumE,
                                                    float* __restrict__ sumW) {
    int i = blockIdx.x * 256 + threadIdx.x;
    if (i >= E_EDGES) return;
    int s = src[i], d = dst[i];
    float e  = lrelu(el[s] + er[d]);
    float z  = __expf(e - fdec(maxE[d]));
    float zw = __expf(w[i] - fdec(maxW[d]));
    atomicAdd(&sumE[d], z);
    atomicAdd(&sumW[d], zw);
}

// ---------------------------------------------------------------------------
// Kernel 4: blended attention + message aggregation. One wave per edge:
// lane holds float4 of ft[src] (b128 gather), 4x global_atomic_add_f32 into
// the L2-resident output (initialized to bias by gat_init).
// ---------------------------------------------------------------------------
__global__ __launch_bounds__(256) void gat_edge_agg(const float* __restrict__ el,
                                                    const float* __restrict__ er,
                                                    const float* __restrict__ w,
                                                    const int* __restrict__ src,
                                                    const int* __restrict__ dst,
                                                    const int* __restrict__ maxE,
                                                    const int* __restrict__ maxW,
                                                    const float* __restrict__ sumE,
                                                    const float* __restrict__ sumW,
                                                    const float* __restrict__ ft,
                                                    float* __restrict__ out) {
    const int wv   = threadIdx.x >> 5;
    const int lane = threadIdx.x & 31;
    const int eidx = blockIdx.x * 8 + wv;
    if (eidx >= E_EDGES) return;

    int s = src[eidx], d = dst[eidx];
    float e  = lrelu(el[s] + er[d]);
    float z  = __expf(e - fdec(maxE[d]));
    float zw = __expf(w[eidx] - fdec(maxW[d]));
    float a  = z * ((1.0f - ALPHA) / sumE[d]) + zw * (ALPHA / sumW[d]);

    const float4 f = reinterpret_cast<const float4*>(ft)[(size_t)s * (OUT_F / 4) + lane];
    float* o = out + (size_t)d * OUT_F + lane * 4;
    atomicAdd(o + 0, f.x * a);
    atomicAdd(o + 1, f.y * a);
    atomicAdd(o + 2, f.z * a);
    atomicAdd(o + 3, f.w * a);
}

// ---------------------------------------------------------------------------
extern "C" void kernel_launch(void* const* d_in, const int* in_sizes, int n_in,
                              void* d_out, int out_size, void* d_ws, size_t ws_size,
                              hipStream_t stream) {
    const float* feat   = (const float*)d_in[0];
    const float* w      = (const float*)d_in[1];
    const float* fcw    = (const float*)d_in[2];
    const float* attn_l = (const float*)d_in[3];
    const float* attn_r = (const float*)d_in[4];
    const float* bias   = (const float*)d_in[5];
    const int*   src    = (const int*)d_in[6];
    const int*   dst    = (const int*)d_in[7];
    float* out = (float*)d_out;

    char* ws = (char*)d_ws;
    size_t o = 0;
    float* ft   = (float*)(ws + o); o += (size_t)N_NODES * OUT_F * sizeof(float);
    float* el   = (float*)(ws + o); o += (size_t)N_NODES * sizeof(float);
    float* er   = (float*)(ws + o); o += (size_t)N_NODES * sizeof(float);
    int*   maxE = (int*)  (ws + o); o += (size_t)N_NODES * sizeof(int);
    int*   maxW = (int*)  (ws + o); o += (size_t)N_NODES * sizeof(int);
    float* sumE = (float*)(ws + o); o += (size_t)N_NODES * sizeof(float);
    float* sumW = (float*)(ws + o); o += (size_t)N_NODES * sizeof(float);

    const int initThreads = N_NODES * OUT_F;
    gat_init<<<(initThreads + 255) / 256, 256, 0, stream>>>(bias, out, maxE, maxW, sumE, sumW);

    gat_gemm<<<(N_NODES + 127) / 128, 256, 0, stream>>>(feat, fcw, attn_l, attn_r, ft, el, er);

    gat_edge_max<<<(E_EDGES + 255) / 256, 256, 0, stream>>>(el, er, w, src, dst, maxE, maxW);

    gat_edge_sum<<<(E_EDGES + 255) / 256, 256, 0, stream>>>(el, er, w, src, dst, maxE, maxW, sumE, sumW);

    gat_edge_agg<<<(E_EDGES + 7) / 8, 256, 0, stream>>>(el, er, w, src, dst, maxE, maxW,
                                                        sumE, sumW, ft, out);
}
